// TopDownBlock_57552561766798
// MI455X (gfx1250) — matrix operand order repaired and checked by reading, not compiled, not run on hardware
//
#include <hip/hip_runtime.h>
#include <cstdint>
#include <cstddef>

// ---------------------------------------------------------------------------
// MI455X (gfx1250) implementation of TopDownBlock.
// All GEMMs run on v_wmma_f32_16x16x32_bf16 (fp32 accumulate), software-
// pipelined so operand loads for step kt+1 overlap the WMMAs of step kt.
// Edge message-passing: W (320x256 bf16 = 160KB) resident in WGP LDS,
// double-buffered A tiles gathered with GLOBAL_LOAD_ASYNC_TO_LDS_B128
// (ASYNCcnt) so the gather overlaps compute; segment_sum via
// global_atomic_add_f32 resolved in L2.
// ---------------------------------------------------------------------------

typedef __attribute__((ext_vector_type(16))) __bf16 v16bf;
typedef __attribute__((ext_vector_type(8)))  float  v8f;

union FragBF { uint4 q[2]; v16bf v; };

// ---- CDNA5 async global->LDS (guarded; falls back to sync copy) -----------
#if defined(__HIP_DEVICE_COMPILE__) &&                                   \
    __has_builtin(__builtin_amdgcn_global_load_async_to_lds_b128) &&      \
    __has_builtin(__builtin_amdgcn_global_load_async_to_lds_b32) &&       \
    __has_builtin(__builtin_amdgcn_s_wait_asynccnt)
#define HAVE_ASYNC_LDS 1
typedef int async_i32x4 __attribute__((ext_vector_type(4)));
typedef __attribute__((address_space(1))) async_i32x4 g_i32x4;   // global vec4
typedef __attribute__((address_space(3))) async_i32x4 l_i32x4;   // LDS vec4
typedef __attribute__((address_space(1))) int g_i32;             // global int
typedef __attribute__((address_space(3))) int l_i32;             // LDS int
#else
#define HAVE_ASYNC_LDS 0
#endif

__device__ __forceinline__ void async_b128(void* lds_dst, const void* gsrc) {
#if HAVE_ASYNC_LDS
  __builtin_amdgcn_global_load_async_to_lds_b128((g_i32x4*)gsrc,
                                                 (l_i32x4*)lds_dst, 0, 0);
#else
  *(uint4*)lds_dst = *(const uint4*)gsrc;
#endif
}

__device__ __forceinline__ void async_b32(void* lds_dst, const void* gsrc) {
#if HAVE_ASYNC_LDS
  __builtin_amdgcn_global_load_async_to_lds_b32((g_i32*)gsrc,
                                                (l_i32*)lds_dst, 0, 0);
#else
  *(int*)lds_dst = *(const int*)gsrc;
#endif
}

__device__ __forceinline__ void async_wait_all() {
#if HAVE_ASYNC_LDS
  __builtin_amdgcn_s_wait_asynccnt(0);
#endif
}

// ---------------------------------------------------------------------------

__device__ __forceinline__ unsigned short f2bf(float f) {
  unsigned int u = __float_as_uint(f);
  u += 0x7FFFu + ((u >> 16) & 1u);          // round-to-nearest-even
  return (unsigned short)(u >> 16);
}

__device__ __forceinline__ float gelu_exact(float x) {
  return 0.5f * x * (1.0f + erff(x * 0.70710678118654752f));
}

// A fragment (16x32 bf16, MxK) from LDS, row stride rs (elements).
// ISA layout: lane&15 = row; lanes>=16 shift K by 8; elems 0..7 -> K kb..kb+7,
// elems 8..15 -> K kb+16..kb+23.
__device__ __forceinline__ v16bf load_a_frag(const unsigned short* A, int rs,
                                             int kt, int lane) {
  int r  = lane & 15;
  int kb = kt * 32 + ((lane >> 4) << 3);
  FragBF f;
  f.q[0] = *(const uint4*)(A + r * rs + kb);
  f.q[1] = *(const uint4*)(A + r * rs + kb + 16);
  return f.v;
}

// B fragment from fragment-major swizzled weights:
// Wz[((kt*NT + nt)*32 + lane)*16 + i]  with  k = kt*32 + (lane>>4)*8 + (i&7) + (i>>3)*16,
//                                           m = nt*16 + (lane&15).
__device__ __forceinline__ v16bf load_b_frag(const unsigned short* Wz, int NT,
                                             int kt, int nt, int lane) {
  const uint4* p = (const uint4*)(Wz + ((((kt * NT + nt) * 32) + lane) << 4));
  FragBF f;
  f.q[0] = p[0];
  f.q[1] = p[1];
  return f.v;
}

// --------------------------- prep kernels ----------------------------------

__global__ void convert_swizzle(const float* __restrict__ W,
                                unsigned short* __restrict__ out,
                                int K, int M) {
  int idx = blockIdx.x * 256 + threadIdx.x;
  if (idx >= K * M) return;
  int NT   = M >> 4;
  int i    = idx & 15;
  int lane = (idx >> 4) & 31;
  int rest = idx >> 9;
  int nt   = rest % NT;
  int kt   = rest / NT;
  int k = kt * 32 + ((lane >> 4) << 3) + (i & 7) + ((i >> 3) << 4);
  int m = nt * 16 + (lane & 15);
  out[idx] = f2bf(W[(size_t)k * M + m]);
}

__global__ void f32_to_bf16(const float* __restrict__ src,
                            unsigned short* __restrict__ dst, int n) {
  int i = blockIdx.x * 256 + threadIdx.x;
  if (i < n) dst[i] = f2bf(src[i]);
}

__global__ void build_xa(const float* __restrict__ x, const float* __restrict__ a,
                         unsigned short* __restrict__ xab, int N) {
  int i = blockIdx.x * 256 + threadIdx.x;
  if (i >= N * 256) return;
  int n = i >> 8, j = i & 255;
  float v = (j < 128) ? x[(size_t)n * 128 + j] : a[(size_t)n * 128 + j - 128];
  xab[i] = f2bf(v);
}

// --------------------------- node GEMM -------------------------------------
// One block (4 waves) per 16-row tile computes all M columns.
// Wave w owns col tiles [w*NTW, (w+1)*NTW); K = KT*32, M = NTW*64.
// Fragments for kt+1 are prefetched before the WMMAs of kt.

template <int KT, int NTW>
__global__ __launch_bounds__(128) void node_linear(
    const unsigned short* __restrict__ Xb,
    const unsigned short* __restrict__ Wz,
    const float* __restrict__ bias,
    float* __restrict__ Yf, unsigned short* __restrict__ Yb,
    const float* __restrict__ addsrc, int act) {
  constexpr int K = KT * 32;
  constexpr int NT = NTW * 4;
  constexpr int M = NT * 16;
  extern __shared__ unsigned char dyn_smem[];
  unsigned short* sA = (unsigned short*)dyn_smem;  // 16 x K bf16
  int tid = threadIdx.x;
  int tile = blockIdx.x;
  {
    uint4* d = (uint4*)sA;
    const uint4* s = (const uint4*)(Xb + (size_t)tile * 16 * K);
#pragma unroll
    for (int c = tid; c < 2 * K; c += 128) async_b128(d + c, s + c);
  }
  async_wait_all();
  __syncthreads();

  int lane = tid & 31;
  int wave = tid >> 5;
  int ntb = wave * NTW;

  v8f acc[NTW] = {};
  v16bf a = load_a_frag(sA, K, 0, lane);
  v16bf b[NTW];
#pragma unroll
  for (int j = 0; j < NTW; ++j) b[j] = load_b_frag(Wz, NT, 0, ntb + j, lane);

#pragma unroll
  for (int kt = 0; kt < KT; ++kt) {
    v16bf an = a;
    v16bf bn[NTW];
#pragma unroll
    for (int j = 0; j < NTW; ++j) bn[j] = b[j];
    if (kt + 1 < KT) {              // prefetch next step's fragments
      an = load_a_frag(sA, K, kt + 1, lane);
#pragma unroll
      for (int j = 0; j < NTW; ++j) bn[j] = load_b_frag(Wz, NT, kt + 1, ntb + j, lane);
    }
#pragma unroll
    for (int j = 0; j < NTW; ++j)
      acc[j] = __builtin_amdgcn_wmma_f32_16x16x32_bf16(false, a, false, b[j],
                                                       (short)0, acc[j], false, false);
    a = an;
#pragma unroll
    for (int j = 0; j < NTW; ++j) b[j] = bn[j];
  }

  int rbase = tile * 16 + ((lane >> 4) << 3);
#pragma unroll
  for (int j = 0; j < NTW; ++j) {
    int col = (ntb + j) * 16 + (lane & 15);
    float bval = bias[col];
#pragma unroll
    for (int i = 0; i < 8; ++i) {
      float v = acc[j][i] + bval;
      if (act) v = gelu_exact(v);
      size_t o = (size_t)(rbase + i) * M + col;
      if (addsrc) v += addsrc[o];
      if (Yf) Yf[o] = v;
      if (Yb) Yb[o] = f2bf(v);
    }
  }
}

// --------------------------- edge message passing ---------------------------
// For each edge tile (16 edges): A = [h_bf16[src] (256) | ef_bf16 (64)],
// msg = gelu(A @ W(320x256) + b); Hacc[dst] += msg (atomic fp32, in-place
// implements h = h + segment_sum).  W resident in LDS; A tiles double-buffered
// and gathered with async global->LDS so the gather overlaps this tile's WMMAs.

#define EMP_W_BYTES  163840                   // 320*256*2
#define EMP_A_BYTES  10240                    // 16*320*2
#define EMP_LDS      (EMP_W_BYTES + 2 * EMP_A_BYTES + 128)

__global__ __launch_bounds__(256) void edge_mp(
    const unsigned short* __restrict__ Hb,    // N x 256 bf16 (snapshot)
    const unsigned short* __restrict__ EFb,   // E x 64 bf16
    const int* __restrict__ srcI, const int* __restrict__ dstI,
    const unsigned short* __restrict__ Wz,    // 320x256 swizzled bf16
    const float* __restrict__ bias,           // 256
    float* __restrict__ Hacc,                 // N x 256 fp32 accumulator
    int ntiles) {
  extern __shared__ unsigned char dyn_smem[];
  unsigned short* sW  = (unsigned short*)dyn_smem;
  unsigned short* sA0 = (unsigned short*)(dyn_smem + EMP_W_BYTES);
  unsigned short* sA1 = sA0 + 16 * 320;
  int* sD0 = (int*)(dyn_smem + EMP_W_BYTES + 2 * EMP_A_BYTES);
  int* sD1 = sD0 + 16;

  int tid = threadIdx.x, lane = tid & 31, wave = tid >> 5;

  {  // stage full W once per block (async: memory -> LDS, no VGPR roundtrip)
    uint4* d = (uint4*)sW;
    const uint4* s = (const uint4*)Wz;
    for (int c = tid; c < 10240; c += 256) async_b128(d + c, s + c);
  }

  // gather one edge tile into LDS: 640 uint4 (per row: 32 from h[src], 8 from
  // ef) + 16 dst indices, all via the async engine.
  auto stage_tile = [&](int t, unsigned short* sA, int* sD) {
    uint4* aq = (uint4*)sA;
#pragma unroll
    for (int u = 0; u < 3; ++u) {
      int c = tid + u * 256;
      if (c < 640) {
        int row = c / 40, j = c - row * 40;
        int e = t * 16 + row;
        const uint4* g = (j < 32)
            ? ((const uint4*)(Hb + (size_t)srcI[e] * 256)) + j
            : ((const uint4*)(EFb + (size_t)e * 64)) + (j - 32);
        async_b128(aq + c, g);
      }
    }
    if (tid < 16) async_b32(sD + tid, dstI + t * 16 + tid);
  };

  const int stride = gridDim.x;
  const int t0 = blockIdx.x;

  // prologue: stage first tile into buffer 0 (W staging still in flight too)
  stage_tile(t0, sA0, sD0);
  async_wait_all();
  __syncthreads();

  int cur = 0;
  for (int t = t0; t < ntiles; t += stride) {
    int tn = t + stride;
    bool hn = tn < ntiles;
    if (hn) stage_tile(tn, cur ? sA0 : sA1, cur ? sD0 : sD1);  // background

    const unsigned short* sA = cur ? sA1 : sA0;
    const int* sD = cur ? sD1 : sD0;

    v8f acc0 = {}, acc1 = {};
    v16bf a  = load_a_frag(sA, 320, 0, lane);
    v16bf b0 = load_b_frag(sW, 16, 0, wave, lane);
    v16bf b1 = load_b_frag(sW, 16, 0, wave + 8, lane);
#pragma unroll
    for (int kt = 0; kt < 10; ++kt) {
      v16bf an = a, b0n = b0, b1n = b1;
      if (kt < 9) {                  // prefetch next step's fragments
        an  = load_a_frag(sA, 320, kt + 1, lane);
        b0n = load_b_frag(sW, 16, kt + 1, wave, lane);
        b1n = load_b_frag(sW, 16, kt + 1, wave + 8, lane);
      }
      acc0 = __builtin_amdgcn_wmma_f32_16x16x32_bf16(false, a, false, b0,
                                                     (short)0, acc0, false, false);
      acc1 = __builtin_amdgcn_wmma_f32_16x16x32_bf16(false, a, false, b1,
                                                     (short)0, acc1, false, false);
      a = an; b0 = b0n; b1 = b1n;
    }

    int cl = lane & 15;
    int rbase = (lane >> 4) << 3;
    float bv0 = bias[wave * 16 + cl];
    float bv1 = bias[(wave + 8) * 16 + cl];
#pragma unroll
    for (int i = 0; i < 8; ++i) {
      int dnode = sD[rbase + i];
      float* rowp = Hacc + (size_t)dnode * 256;
      unsafeAtomicAdd(rowp + wave * 16 + cl, gelu_exact(acc0[i] + bv0));
      unsafeAtomicAdd(rowp + (wave + 8) * 16 + cl, gelu_exact(acc1[i] + bv1));
    }

    async_wait_all();   // next tile's gather landed
    __syncthreads();
    cur ^= 1;
  }
}

// --------------------------- latent / residual ------------------------------

__global__ void latent_kernel(const float* __restrict__ pfeat,
                              const float* __restrict__ dmv,
                              const float* __restrict__ eps,
                              float* __restrict__ kl,
                              unsigned short* __restrict__ zsb, int N) {
  int i = blockIdx.x * 256 + threadIdx.x;
  if (i >= N * 64) return;
  int n = i >> 6, j = i & 63;
  float pm = pfeat[(size_t)n * 192 + j];
  float pv = pfeat[(size_t)n * 192 + 64 + j];
  float dm = dmv[(size_t)n * 128 + j];
  float dv = dmv[(size_t)n * 128 + 64 + j];
  float qv = pv + dv;
  float qm = pm + dm;
  float eqv = expf(qv), epv = expf(pv);
  float zs = eqv * eps[i] + qm;
  kl[i] = -0.5f + pv - qv + 0.5f * (eqv * eqv + dm * dm) / (epv * epv);
  zsb[i] = f2bf(zs);
}

__global__ void resin_kernel(const float* __restrict__ x,
                             const float* __restrict__ pfeat,
                             const float* __restrict__ zf,
                             float* __restrict__ xrf,
                             unsigned short* __restrict__ xrb, int N) {
  int i = blockIdx.x * 256 + threadIdx.x;
  if (i >= N * 128) return;
  int n = i >> 7, j = i & 127;
  float v = x[i] + pfeat[(size_t)n * 192 + 128 + j] + zf[i];
  xrf[i] = v;
  xrb[i] = f2bf(v);
}

// ------------------------------ launch --------------------------------------

extern "C" void kernel_launch(void* const* d_in, const int* in_sizes, int n_in,
                              void* d_out, int out_size, void* d_ws, size_t ws_size,
                              hipStream_t stream) {
  (void)n_in; (void)out_size; (void)ws_size;
  const int N = in_sizes[0] / 128;
  const int E = in_sizes[3] / 64;
  const float* x   = (const float*)d_in[0];
  const float* a   = (const float*)d_in[1];
  const int*   ei  = (const int*)d_in[2];
  const float* ef  = (const float*)d_in[3];
  const float* eps = (const float*)d_in[4];
  const int* srcI = ei;
  const int* dstI = ei + E;

  // weights in d_in: prior(ff1,mp1,mp2,ff2) post(...) res(...) zff1 zff2
  static const int wIdx[14] = {5,7,9,11, 13,15,17,19, 21,23,25,27, 29,31};
  static const int wK[14]   = {128,320,320,256, 256,320,320,256, 128,320,320,256, 64,128};
  static const int wM[14]   = {256,256,256,192, 256,256,256,128, 256,256,256,128, 128,128};
  const float* bb[14];
  for (int i = 0; i < 14; ++i) bb[i] = (const float*)d_in[wIdx[i] + 1];

  char* ws = (char*)d_ws;
  size_t off = 0;
  auto walloc = [&](size_t bytes) -> char* {
    off = (off + 255) & ~(size_t)255;
    char* p = ws + off;
    off += bytes;
    return p;
  };
  unsigned short* wz[14];
  for (int i = 0; i < 14; ++i) wz[i] = (unsigned short*)walloc((size_t)wK[i] * wM[i] * 2);
  unsigned short* efb = (unsigned short*)walloc((size_t)E * 64 * 2);
  unsigned short* xb  = (unsigned short*)walloc((size_t)N * 128 * 2);
  unsigned short* xab = (unsigned short*)walloc((size_t)N * 256 * 2);
  float*          hF  = (float*)walloc((size_t)N * 256 * 4);
  unsigned short* hB  = (unsigned short*)walloc((size_t)N * 256 * 2);
  float*          pfe = (float*)walloc((size_t)N * 192 * 4);
  float*          dmv = (float*)walloc((size_t)N * 128 * 4);
  unsigned short* zsb = (unsigned short*)walloc((size_t)N * 64 * 2);
  unsigned short* tb  = (unsigned short*)walloc((size_t)N * 128 * 2);
  float*          zf  = (float*)walloc((size_t)N * 128 * 4);
  float*          xrf = (float*)walloc((size_t)N * 128 * 4);
  unsigned short* xrb = (unsigned short*)walloc((size_t)N * 128 * 2);

  float* outX  = (float*)d_out;
  float* outKL = (float*)d_out + (size_t)N * 128;

  // prep: weights + activations -> bf16
  for (int i = 0; i < 14; ++i) {
    int tot = wK[i] * wM[i];
    convert_swizzle<<<dim3((tot + 255) / 256), dim3(256), 0, stream>>>(
        (const float*)d_in[wIdx[i]], wz[i], wK[i], wM[i]);
  }
  f32_to_bf16<<<dim3((E * 64 + 255) / 256), dim3(256), 0, stream>>>(ef, efb, E * 64);
  f32_to_bf16<<<dim3((N * 128 + 255) / 256), dim3(256), 0, stream>>>(x, xb, N * 128);
  build_xa<<<dim3((N * 256 + 255) / 256), dim3(256), 0, stream>>>(x, a, xab, N);

  const int tilesN = N / 16;
  const int tilesE = E / 16;
  dim3 b256(256);
  dim3 egrid(1024);
  int refN = (N * 256 + 255) / 256;

  // node_linear dispatch helper (KT = K/32, NTW = M/64)
  auto nl = [&](int KT, int NTW, const unsigned short* Xb, const unsigned short* Wzp,
                const float* bias, float* Yf, unsigned short* Yb,
                const float* addsrc, int act) {
    size_t sh = (size_t)16 * (KT * 32) * 2;
    dim3 g(tilesN), b(128);
    if (KT == 4 && NTW == 4)      node_linear<4, 4><<<g, b, sh, stream>>>(Xb, Wzp, bias, Yf, Yb, addsrc, act);
    else if (KT == 8 && NTW == 3) node_linear<8, 3><<<g, b, sh, stream>>>(Xb, Wzp, bias, Yf, Yb, addsrc, act);
    else if (KT == 8 && NTW == 4) node_linear<8, 4><<<g, b, sh, stream>>>(Xb, Wzp, bias, Yf, Yb, addsrc, act);
    else if (KT == 8 && NTW == 2) node_linear<8, 2><<<g, b, sh, stream>>>(Xb, Wzp, bias, Yf, Yb, addsrc, act);
    else if (KT == 2 && NTW == 2) node_linear<2, 2><<<g, b, sh, stream>>>(Xb, Wzp, bias, Yf, Yb, addsrc, act);
    else if (KT == 4 && NTW == 2) node_linear<4, 2><<<g, b, sh, stream>>>(Xb, Wzp, bias, Yf, Yb, addsrc, act);
  };

  // ---- prior block ----
  nl(4, 4, xb, wz[0], bb[0], hF, hB, nullptr, 1);
  edge_mp<<<egrid, b256, EMP_LDS, stream>>>(hB, efb, srcI, dstI, wz[1], bb[1], hF, tilesE);
  f32_to_bf16<<<dim3(refN), b256, 0, stream>>>(hF, hB, N * 256);
  edge_mp<<<egrid, b256, EMP_LDS, stream>>>(hB, efb, srcI, dstI, wz[2], bb[2], hF, tilesE);
  f32_to_bf16<<<dim3(refN), b256, 0, stream>>>(hF, hB, N * 256);
  nl(8, 3, hB, wz[3], bb[3], pfe, nullptr, nullptr, 0);

  // ---- post block ----
  nl(8, 4, xab, wz[4], bb[4], hF, hB, nullptr, 1);
  edge_mp<<<egrid, b256, EMP_LDS, stream>>>(hB, efb, srcI, dstI, wz[5], bb[5], hF, tilesE);
  f32_to_bf16<<<dim3(refN), b256, 0, stream>>>(hF, hB, N * 256);
  edge_mp<<<egrid, b256, EMP_LDS, stream>>>(hB, efb, srcI, dstI, wz[6], bb[6], hF, tilesE);
  f32_to_bf16<<<dim3(refN), b256, 0, stream>>>(hF, hB, N * 256);
  nl(8, 2, hB, wz[7], bb[7], dmv, nullptr, nullptr, 0);

  // ---- latent head ----
  latent_kernel<<<dim3((N * 64 + 255) / 256), b256, 0, stream>>>(pfe, dmv, eps, outKL, zsb, N);
  nl(2, 2, zsb, wz[12], bb[12], nullptr, tb, nullptr, 1);
  nl(4, 2, tb, wz[13], bb[13], zf, nullptr, nullptr, 0);
  resin_kernel<<<dim3((N * 128 + 255) / 256), b256, 0, stream>>>(x, pfe, zf, xrf, xrb, N);

  // ---- residual block ----
  nl(4, 4, xrb, wz[8], bb[8], hF, hB, nullptr, 1);
  edge_mp<<<egrid, b256, EMP_LDS, stream>>>(hB, efb, srcI, dstI, wz[9], bb[9], hF, tilesE);
  f32_to_bf16<<<dim3(refN), b256, 0, stream>>>(hF, hB, N * 256);
  edge_mp<<<egrid, b256, EMP_LDS, stream>>>(hB, efb, srcI, dstI, wz[10], bb[10], hF, tilesE);
  f32_to_bf16<<<dim3(refN), b256, 0, stream>>>(hF, hB, N * 256);
  nl(8, 2, hB, wz[11], bb[11], outX, nullptr, xrf, 0);
}
